// SelfContrastiveLoss_54296976556236
// MI455X (gfx1250) — compile-verified, matrix operand order checked by reading, not run on hardware
//
#include <hip/hip_runtime.h>
#include <hip/hip_bf16.h>

typedef __attribute__((ext_vector_type(16))) _Float16 v16h;
typedef __attribute__((ext_vector_type(8)))  _Float16 v8h;
typedef __attribute__((ext_vector_type(8)))  float    v8f;

#define N_ROWS 8192
#define DIM    256
#define NSEG   16
#define TILES_PER_SEG (N_ROWS / 16 / NSEG)   // 512 col-tiles / 16 segments = 32
#define LDS_PITCH 264                        // 256 halves + 8 pad (528B = 132 dwords -> 4-bank rotation/row)
#define ROWS_PER_BLOCK 256                   // 8 waves x 32 rows

struct U32x8 { uint4 lo; uint4 hi; };

// ---------------- Kernel 1: row-normalize (fold sqrt(1/TEMP)=sqrt(2)) + label compaction ----------------
__global__ __launch_bounds__(256) void norm_kernel(const float* __restrict__ emb,
                                                   const int* __restrict__ label64,
                                                   _Float16* __restrict__ normed,
                                                   int* __restrict__ lab32) {
    int wave = threadIdx.x >> 5;
    int lane = threadIdx.x & 31;
    int row  = blockIdx.x * 8 + wave;
    const float* src = emb + (size_t)row * DIM + lane * 8;
    float4 x0 = *(const float4*)(src);
    float4 x1 = *(const float4*)(src + 4);
    float ss = x0.x*x0.x + x0.y*x0.y + x0.z*x0.z + x0.w*x0.w
             + x1.x*x1.x + x1.y*x1.y + x1.z*x1.z + x1.w*x1.w;
    #pragma unroll
    for (int off = 16; off >= 1; off >>= 1) ss += __shfl_xor(ss, off, 32);
    float scale = 1.41421356237f / sqrtf(ss);   // includes 1/sqrt(TEMP): dot gives sim/TEMP directly
    v8h o;
    o[0] = (_Float16)(x0.x*scale); o[1] = (_Float16)(x0.y*scale);
    o[2] = (_Float16)(x0.z*scale); o[3] = (_Float16)(x0.w*scale);
    o[4] = (_Float16)(x1.x*scale); o[5] = (_Float16)(x1.y*scale);
    o[6] = (_Float16)(x1.z*scale); o[7] = (_Float16)(x1.w*scale);
    *(v8h*)(normed + (size_t)row * DIM + lane * 8) = o;

    int gid = blockIdx.x * 256 + threadIdx.x;
    if (gid < N_ROWS) lab32[gid] = label64[2 * gid];   // int64 -> int32 (labels < 100)
}

// ---------------- Kernel 2: fused WMMA tile pass, 32 rows per wave ----------------
// grid.x = N/256 row blocks (8 waves x 32 rows), grid.y = NSEG column segments.
__global__ __launch_bounds__(256) void tile_kernel(const _Float16* __restrict__ normed,
                                                   const int* __restrict__ lab32,
                                                   float* __restrict__ pSum,
                                                   float* __restrict__ pPos,
                                                   float* __restrict__ pCnt) {
    __shared__ __align__(16) _Float16 bt[2][16][LDS_PITCH];

    const int tid     = threadIdx.x;
    const int wave    = tid >> 5;
    const int lane    = tid & 31;
    const int r16     = lane & 15;          // N index in C; row-within-tile for frag loads
    const int halfSel = lane >> 4;          // 0: lanes 0-15, 1: lanes 16-31
    const int kbase   = halfSel * 8;        // K offset of this half-wave within a 32-wide K step
    const int rowBase = blockIdx.x * ROWS_PER_BLOCK + wave * 32;   // two 16-row tiles
    const int seg     = blockIdx.y;

    // Resident A fragments for BOTH 16-row tiles: 2 x (8 chunks of 16x32 f16) = 128 VGPRs.
    v16h afrag[2][8];
    #pragma unroll
    for (int h = 0; h < 2; ++h) {
        const _Float16* arow = normed + (size_t)(rowBase + h * 16 + r16) * DIM;
        #pragma unroll
        for (int c = 0; c < 8; ++c) {
            U32x8 u;
            u.lo = *(const uint4*)(arow + c * 32 + kbase);        // K = c*32+kbase .. +7
            u.hi = *(const uint4*)(arow + c * 32 + kbase + 16);   // K = c*32+kbase+16 .. +23
            afrag[h][c] = __builtin_bit_cast(v16h, u);
        }
    }

    // Row labels: M = r + 8*halfSel within each 16-row tile half.
    int rlab[2][8];
    #pragma unroll
    for (int h = 0; h < 2; ++h)
        #pragma unroll
        for (int r = 0; r < 8; ++r)
            rlab[h][r] = lab32[rowBase + h * 16 + r + halfSel * 8];

    float accS[2][8], accP[2][8], accC[2][8];
    #pragma unroll
    for (int h = 0; h < 2; ++h)
        #pragma unroll
        for (int r = 0; r < 8; ++r) { accS[h][r] = 0.f; accP[h][r] = 0.f; accC[h][r] = 0.f; }

    auto fillB = [&](int buf, int colBase) {
        #pragma unroll
        for (int i = 0; i < 2; ++i) {
            int idx  = tid + i * 256;   // 0..511 uint4 chunks (16 rows x 32 chunks)
            int brow = idx >> 5;
            int c16  = idx & 31;
            uint4 v = *(const uint4*)(normed + (size_t)(colBase + brow) * DIM + c16 * 8);
            *(uint4*)(&bt[buf][brow][c16 * 8]) = v;
        }
    };

    const int colTile0 = seg * TILES_PER_SEG;
    fillB(0, colTile0 * 16);
    __syncthreads();

    for (int t = 0; t < TILES_PER_SEG; ++t) {
        const int cur = t & 1;
        if (t + 1 < TILES_PER_SEG) fillB(cur ^ 1, (colTile0 + t + 1) * 16);

        const int colBase = (colTile0 + t) * 16;
        const int lj = lab32[colBase + r16];   // hoisted: hides under the WMMA chain

        // Each B fragment pair feeds TWO WMMAs (both row halves): halves LDS traffic per matrix op.
        v8f cacc0 = {}, cacc1 = {};
        #pragma unroll
        for (int c = 0; c < 8; ++c) {
            U32x8 u;
            u.lo = *(const uint4*)(&bt[cur][r16][c * 32 + kbase]);
            u.hi = *(const uint4*)(&bt[cur][r16][c * 32 + kbase + 16]);
            v16h bfrag = __builtin_bit_cast(v16h, u);
            cacc0 = __builtin_amdgcn_wmma_f32_16x16x32_f16(
                false, afrag[0][c], false, bfrag, (short)0, cacc0, false, false);
            cacc1 = __builtin_amdgcn_wmma_f32_16x16x32_f16(
                false, afrag[1][c], false, bfrag, (short)0, cacc1, false, false);
        }

        // Post-process each 16-row half; diagonal only when colBase == half's rowBase
        // (wave-uniform -> force scalar branch via readfirstlane).
        #pragma unroll
        for (int h = 0; h < 2; ++h) {
            const v8f& cc = h ? cacc1 : cacc0;
            const int rb = rowBase + h * 16;
            if (__builtin_amdgcn_readfirstlane(colBase != rb)) {
                // Hot path: no diagonal element in this tile.
                #pragma unroll
                for (int r = 0; r < 8; ++r) {
                    const float s = cc[r];
                    accS[h][r] += __expf(s);
                    const float m = (lj == rlab[h][r]) ? 1.0f : 0.0f;
                    accP[h][r] = fmaf(m, s, accP[h][r]);
                    accC[h][r] += m;
                }
            } else {
                // Diagonal tile (once per half per sweep): i==j iff r16 == r + 8*halfSel.
                #pragma unroll
                for (int r = 0; r < 8; ++r) {
                    const float s = cc[r];
                    const bool diag = (r16 == r + halfSel * 8);
                    const float e = __expf(s);
                    accS[h][r] += diag ? 0.0f : e;
                    const float m = (!diag && (lj == rlab[h][r])) ? 1.0f : 0.0f;
                    accP[h][r] = fmaf(m, s, accP[h][r]);
                    accC[h][r] += m;
                }
            }
        }
        __syncthreads();
    }

    // Reduce over the 16 lanes that share each M row (N dimension), then write partials.
    #pragma unroll
    for (int h = 0; h < 2; ++h) {
        #pragma unroll
        for (int r = 0; r < 8; ++r) {
            #pragma unroll
            for (int off = 8; off >= 1; off >>= 1) {
                accS[h][r] += __shfl_xor(accS[h][r], off, 16);
                accP[h][r] += __shfl_xor(accP[h][r], off, 16);
                accC[h][r] += __shfl_xor(accC[h][r], off, 16);
            }
        }
    }
    if (r16 == 0) {
        #pragma unroll
        for (int h = 0; h < 2; ++h)
            #pragma unroll
            for (int r = 0; r < 8; ++r) {
                int row = rowBase + h * 16 + r + halfSel * 8;
                pSum[seg * N_ROWS + row] = accS[h][r];
                pPos[seg * N_ROWS + row] = accP[h][r];
                pCnt[seg * N_ROWS + row] = accC[h][r];
            }
    }
}

// ---------------- Kernel 3: finalize + scalar reduce ----------------
__global__ __launch_bounds__(1024) void final_kernel(const float* __restrict__ pSum,
                                                     const float* __restrict__ pPos,
                                                     const float* __restrict__ pCnt,
                                                     float* __restrict__ out) {
    __shared__ float red[1024];
    int tid = threadIdx.x;
    float tot = 0.f;
    for (int i = tid; i < N_ROWS; i += 1024) {
        float s = 0.f, p = 0.f, c = 0.f;
        #pragma unroll
        for (int g = 0; g < NSEG; ++g) {
            s += pSum[g * N_ROWS + i];
            p += pPos[g * N_ROWS + i];
            c += pCnt[g * N_ROWS + i];
        }
        if (c > 0.f) tot += logf(s) - p / c;
    }
    red[tid] = tot;
    __syncthreads();
    for (int st = 512; st >= 1; st >>= 1) {
        if (tid < st) red[tid] += red[tid + st];
        __syncthreads();
    }
    if (tid == 0) out[0] = red[0];
}

extern "C" void kernel_launch(void* const* d_in, const int* in_sizes, int n_in,
                              void* d_out, int out_size, void* d_ws, size_t ws_size,
                              hipStream_t stream) {
    const float* emb   = (const float*)d_in[0];
    const int*   lab64 = (const int*)d_in[1];   // int64 labels, read low 32 bits
    float*       out   = (float*)d_out;

    _Float16* normed = (_Float16*)d_ws;
    char* p = (char*)d_ws + (size_t)N_ROWS * DIM * sizeof(_Float16);
    int*   lab32 = (int*)p;                 p += (size_t)N_ROWS * sizeof(int);
    float* pSum  = (float*)p;               p += (size_t)NSEG * N_ROWS * sizeof(float);
    float* pPos  = (float*)p;               p += (size_t)NSEG * N_ROWS * sizeof(float);
    float* pCnt  = (float*)p;

    norm_kernel<<<N_ROWS / 8, 256, 0, stream>>>(emb, lab64, normed, lab32);

    dim3 grid(N_ROWS / ROWS_PER_BLOCK, NSEG);
    tile_kernel<<<grid, 256, 0, stream>>>(normed, lab32, pSum, pPos, pCnt);

    final_kernel<<<1, 1024, 0, stream>>>(pSum, pPos, pCnt, out);
}
